// BoxMultiHeadedAttention_13597866459935
// MI455X (gfx1250) — matrix operand order repaired and checked by reading, not compiled, not run on hardware
//
#include <hip/hip_runtime.h>

typedef __attribute__((ext_vector_type(16))) _Float16 v16h;
typedef __attribute__((ext_vector_type(8)))  float    v8f;

#define BB  4
#define NN  512
#define DDIM 512
#define HH  8
#define DKK 64

// ---------------------------------------------------------------------------
// helpers
// ---------------------------------------------------------------------------
__device__ __forceinline__ void lds_wave_fence() {
  // wave-local LDS RAW fence (CDNA5 split counters)
  asm volatile("s_wait_dscnt 0" ::: "memory");
}

__device__ __forceinline__ unsigned pack2(_Float16 lo, _Float16 hi) {
  union { _Float16 h[2]; unsigned u; } x;
  x.h[0] = lo; x.h[1] = hi;
  return x.u;
}

// A-matrix fragment, 16x32 f16, row-major source (stride = f16 elements/row)
// lane m=lane&15; K = 16*half+e (contiguous, matches B fragments)
// -> 16 contiguous f16 per lane => ds_load_b128 pair.
__device__ __forceinline__ v16h frag_a_16x32(const _Float16* base, int stride, int lane) {
  const int half = (lane >> 4) & 1;
  const int m = lane & 15;
  v16h a;
#pragma unroll
  for (int e = 0; e < 16; ++e) {
    const int k = 16 * half + e;
    a[e] = base[m * stride + k];
  }
  return a;
}

// B-matrix fragment (32x16 K x N) read from an N-major (transposed) tile
// T[N][K] with row stride = K elems. Lane n=lane&15 needs K=16*half+e
// -> 16 contiguous f16 per lane => ds_load_b128 pair.
__device__ __forceinline__ v16h frag_b_from_T(const _Float16* baseT, int stride, int lane) {
  const int half = (lane >> 4) & 1;
  const int n = lane & 15;
  v16h b;
#pragma unroll
  for (int e = 0; e < 16; ++e)
    b[e] = baseT[n * stride + 16 * half + e];
  return b;
}

// ---------------------------------------------------------------------------
// f32 -> f16 convert
// ---------------------------------------------------------------------------
__global__ void cvt_f16_kernel(const float* __restrict__ src,
                               _Float16* __restrict__ dst, int n) {
  int i = blockIdx.x * blockDim.x + threadIdx.x;
  if (i < n) dst[i] = (_Float16)src[i];
}

// ---------------------------------------------------------------------------
// WMMA GEMM: Y[M][Nout] = A[M][K] @ W[K][Nout] + bias, f16 in, f32 acc.
// Output f16 (Yh) or f32 (Yf). 128 threads = 4 waves, 64x64 tile.
// W tile staged TRANSPOSED in LDS so B-fragment reads are contiguous.
// ---------------------------------------------------------------------------
__global__ __launch_bounds__(128) void gemm_wmma_kernel(
    const _Float16* __restrict__ A, const _Float16* __restrict__ W,
    const float* __restrict__ bias, _Float16* __restrict__ Yh,
    float* __restrict__ Yf, int M, int K, int Nout) {
  __shared__ _Float16 At[64][32];    // row-major 64 x 32
  __shared__ _Float16 BtT[64][32];   // transposed: [n][k] 64 x 32

  const int tid = threadIdx.x;
  const int lane = tid & 31;
  const int wave = tid >> 5;
  const int n0 = blockIdx.x * 64;
  const int m0 = blockIdx.y * 64;

  v8f acc[4] = {{0,0,0,0,0,0,0,0},{0,0,0,0,0,0,0,0},
                {0,0,0,0,0,0,0,0},{0,0,0,0,0,0,0,0}};

  for (int kk = 0; kk < K; kk += 32) {
    {  // A tile 64x32 : 128 thr * 16 f16, direct copy
      const int r = tid >> 1, cb = (tid & 1) * 16;
      const uint4* s = (const uint4*)(A + (size_t)(m0 + r) * K + kk + cb);
      uint4* d = (uint4*)(&At[r][cb]);
      d[0] = s[0]; d[1] = s[1];
    }
    {  // W tile 32x64, transposed into BtT: thread takes k-pair x 8 columns
      const int kp = tid >> 3;           // 0..15 -> k = 2kp, 2kp+1
      const int nb = (tid & 7) * 8;      // 8 columns
      const uint4 r0 = *(const uint4*)(W + (size_t)(kk + 2 * kp) * Nout + n0 + nb);
      const uint4 r1 = *(const uint4*)(W + (size_t)(kk + 2 * kp + 1) * Nout + n0 + nb);
      const _Float16* h0 = (const _Float16*)&r0;
      const _Float16* h1 = (const _Float16*)&r1;
#pragma unroll
      for (int d = 0; d < 8; ++d)
        *(unsigned*)&BtT[nb + d][2 * kp] = pack2(h0[d], h1[d]);
    }
    __syncthreads();
    const v16h a = frag_a_16x32(&At[wave * 16][0], 32, lane);
#pragma unroll
    for (int j = 0; j < 4; ++j) {
      const v16h b = frag_b_from_T(&BtT[j * 16][0], 32, lane);
      acc[j] = __builtin_amdgcn_wmma_f32_16x16x32_f16(
          false, a, false, b, (short)0, acc[j], false, false);
    }
    __syncthreads();
  }

  const int half = lane >> 4, nloc = lane & 15;
  const int mrow = m0 + wave * 16 + 8 * half;
#pragma unroll
  for (int j = 0; j < 4; ++j) {
    const int col = n0 + j * 16 + nloc;
    const float bv = bias[col];
#pragma unroll
    for (int r = 0; r < 8; ++r) {
      const float v = acc[j][r] + bv;
      const size_t idx = (size_t)(mrow + r) * Nout + col;
      if (Yh) Yh[idx] = (_Float16)v;
      else    Yf[idx] = v;
    }
  }
}

// ---------------------------------------------------------------------------
// Geometry logits with mask folded in:
// logg[b][h][i][j] = log(clip(relu(emb(i,j).Wg[:,h]+bg[h]),1e-6))
//                    + (mask[b][i][j]==0 ? -1e9 : 0)
// ---------------------------------------------------------------------------
__global__ __launch_bounds__(256) void geo_kernel(
    const float* __restrict__ box, const float* __restrict__ Wg,
    const float* __restrict__ bg, const int* __restrict__ mask,
    float* __restrict__ logg) {
  __shared__ float sWg[64][HH];
  __shared__ float sbg[HH];
  const int tid = threadIdx.x;
  if (tid < 64) {
#pragma unroll
    for (int h = 0; h < HH; ++h) sWg[tid][h] = Wg[tid * HH + h];
  }
  if (tid < HH) sbg[tid] = bg[tid];

  const int bi = blockIdx.x;           // b*N + i
  const int b = bi >> 9;
  const int i = bi & (NN - 1);
  const float xi0 = box[bi * 4 + 0], xi1 = box[bi * 4 + 1];
  const float yi0 = box[bi * 4 + 2], yi1 = box[bi * 4 + 3];
  const float cxi = 0.5f * (xi0 + xi1), cyi = 0.5f * (yi0 + yi1);
  const float wi = xi1 - xi0 + 1.0f, hi = yi1 - yi0 + 1.0f;
  __syncthreads();

  for (int j = tid; j < NN; j += 256) {
    const int bj = b * NN + j;
    const float x0 = box[bj * 4 + 0], x1 = box[bj * 4 + 1];
    const float y0 = box[bj * 4 + 2], y1 = box[bj * 4 + 3];
    const float cxj = 0.5f * (x0 + x1), cyj = 0.5f * (y0 + y1);
    const float wj = x1 - x0 + 1.0f, hj = y1 - y0 + 1.0f;
    float pos[4];
    pos[0] = __logf(fmaxf(fabsf((cxi - cxj) / wi), 1e-3f));
    pos[1] = __logf(fmaxf(fabsf((cyi - cyj) / hi), 1e-3f));
    pos[2] = __logf(wi / wj);
    pos[3] = __logf(hi / hj);

    const float mterm =
        (mask[((size_t)b * NN + i) * NN + j] == 0) ? -1e9f : 0.0f;

    float acc[HH];
#pragma unroll
    for (int h = 0; h < HH; ++h) acc[h] = sbg[h];
#pragma unroll
    for (int p = 0; p < 4; ++p) {
#pragma unroll
      for (int f = 0; f < 8; ++f) {
        const float dm = __expf(-0.8634693f * (float)f);  // 1000^(-f/8)
        const float ang = 100.0f * pos[p] * dm;
        const float sv = __sinf(ang), cv = __cosf(ang);
        const int g = p * 8 + f;
#pragma unroll
        for (int h = 0; h < HH; ++h)
          acc[h] += sv * sWg[g][h] + cv * sWg[32 + g][h];
      }
    }
#pragma unroll
    for (int h = 0; h < HH; ++h) {
      float v = fmaxf(acc[h], 0.0f);
      v = __logf(fmaxf(v, 1e-6f)) + mterm;
      logg[(((size_t)b * HH + h) * NN + i) * NN + j] = v;
    }
  }
}

// ---------------------------------------------------------------------------
// Flash attention: 256 blocks = (b, h, 64-row q band); 4 waves, one 16-row
// q-tile each. K tile row-major (K^T gather is contiguous); V tile staged
// TRANSPOSED so the P.V B-fragment reads are contiguous.
// ---------------------------------------------------------------------------
__global__ __launch_bounds__(128) void attn_kernel(
    const _Float16* __restrict__ Qh, const _Float16* __restrict__ Kh,
    const _Float16* __restrict__ Vh, const float* __restrict__ logg,
    _Float16* __restrict__ Oh) {
  __shared__ _Float16 Qt[64][64];
  __shared__ _Float16 Kt[32][64];    // [key][dk]
  __shared__ _Float16 VtT[64][32];   // transposed: [dk][key]
  __shared__ _Float16 Pt[4][16][32];

  const int tid = threadIdx.x, lane = tid & 31, wave = tid >> 5;
  const int blk = blockIdx.x;
  const int qb = blk & 7;
  const int h = (blk >> 3) & 7;
  const int b = blk >> 6;
  const int q0 = qb * 64;

  {  // Q tile 64x64 : 128 thr * 32 f16
    const int r = tid >> 1, cb = (tid & 1) * 32;
    const uint4* s =
        (const uint4*)(Qh + ((size_t)(b * NN + q0 + r)) * DDIM + h * DKK + cb);
    uint4* d = (uint4*)(&Qt[r][cb]);
    d[0] = s[0]; d[1] = s[1]; d[2] = s[2]; d[3] = s[3];
  }
  __syncthreads();

  const int half = lane >> 4, nloc = lane & 15;
  const v16h qa0 = frag_a_16x32(&Qt[wave * 16][0], 64, lane);
  const v16h qa1 = frag_a_16x32(&Qt[wave * 16][32], 64, lane);

  v8f o[4] = {{0,0,0,0,0,0,0,0},{0,0,0,0,0,0,0,0},
              {0,0,0,0,0,0,0,0},{0,0,0,0,0,0,0,0}};
  float mrow[8], lrow[8];
#pragma unroll
  for (int r = 0; r < 8; ++r) { mrow[r] = -1e30f; lrow[r] = 0.0f; }

  const int qrowbase = q0 + wave * 16 + 8 * half;
  const float scale = 0.125f;  // 1/sqrt(64)

  for (int k0 = 0; k0 < NN; k0 += 32) {
    {  // K tile 32x64 row-major, direct copy
      const int r = tid >> 2, cb = (tid & 3) * 16;
      const uint4* ks =
          (const uint4*)(Kh + ((size_t)(b * NN + k0 + r)) * DDIM + h * DKK + cb);
      uint4* kd = (uint4*)(&Kt[r][cb]);
      kd[0] = ks[0]; kd[1] = ks[1];
    }
    {  // V tile 32x64 -> transposed VtT[dk][key]; thread = key-pair x 8 dk
      const int kp = tid >> 3;            // 0..15 -> key = 2kp, 2kp+1
      const int db = (tid & 7) * 8;       // 8 dk values
      const size_t row0 = ((size_t)(b * NN + k0 + 2 * kp)) * DDIM + h * DKK + db;
      const uint4 r0 = *(const uint4*)(Vh + row0);
      const uint4 r1 = *(const uint4*)(Vh + row0 + DDIM);
      const _Float16* h0 = (const _Float16*)&r0;
      const _Float16* h1 = (const _Float16*)&r1;
#pragma unroll
      for (int d = 0; d < 8; ++d)
        *(unsigned*)&VtT[db + d][2 * kp] = pack2(h0[d], h1[d]);
    }
    __syncthreads();

    // S[16q x 32keys] as two C fragments; B = K^T gathered contiguous from Kt
    v8f s[2];
#pragma unroll
    for (int c = 0; c < 2; ++c) {
      v16h bk0, bk1;
#pragma unroll
      for (int e = 0; e < 16; ++e) {
        bk0[e] = Kt[16 * c + nloc][16 * half + e];
        bk1[e] = Kt[16 * c + nloc][32 + 16 * half + e];
      }
      v8f sc = {0,0,0,0,0,0,0,0};
      sc = __builtin_amdgcn_wmma_f32_16x16x32_f16(false, qa0, false, bk0,
                                                  (short)0, sc, false, false);
      sc = __builtin_amdgcn_wmma_f32_16x16x32_f16(false, qa1, false, bk1,
                                                  (short)0, sc, false, false);
      s[c] = sc;
    }

    // logits = scores*scale + (geo log-weights with mask folded in)
    float t[2][8];
#pragma unroll
    for (int c = 0; c < 2; ++c) {
      const int key = k0 + 16 * c + nloc;
#pragma unroll
      for (int r = 0; r < 8; ++r) {
        const int q = qrowbase + r;
        t[c][r] = s[c][r] * scale +
                  logg[(((size_t)b * HH + h) * NN + q) * NN + key];
      }
    }

    // online softmax: row max over 32 keys (pairwise + 16-lane butterfly)
    float rmax[8];
#pragma unroll
    for (int r = 0; r < 8; ++r) rmax[r] = fmaxf(t[0][r], t[1][r]);
#pragma unroll
    for (int off = 8; off >= 1; off >>= 1)
#pragma unroll
      for (int r = 0; r < 8; ++r)
        rmax[r] = fmaxf(rmax[r], __shfl_xor(rmax[r], off, 16));

    float alpha[8];
#pragma unroll
    for (int r = 0; r < 8; ++r) {
      const float mnew = fmaxf(mrow[r], rmax[r]);
      alpha[r] = __expf(mrow[r] - mnew);
      mrow[r] = mnew;
    }
#pragma unroll
    for (int c = 0; c < 2; ++c)
#pragma unroll
      for (int r = 0; r < 8; ++r) t[c][r] = __expf(t[c][r] - mrow[r]);

    float rsum[8];
#pragma unroll
    for (int r = 0; r < 8; ++r) rsum[r] = t[0][r] + t[1][r];
#pragma unroll
    for (int off = 8; off >= 1; off >>= 1)
#pragma unroll
      for (int r = 0; r < 8; ++r) rsum[r] += __shfl_xor(rsum[r], off, 16);
#pragma unroll
    for (int r = 0; r < 8; ++r) lrow[r] = lrow[r] * alpha[r] + rsum[r];
#pragma unroll
    for (int j = 0; j < 4; ++j)
#pragma unroll
      for (int r = 0; r < 8; ++r) o[j][r] *= alpha[r];

    // stage P (C layout -> f16 LDS) and regather as A fragment
#pragma unroll
    for (int c = 0; c < 2; ++c)
#pragma unroll
      for (int r = 0; r < 8; ++r)
        Pt[wave][8 * half + r][16 * c + nloc] = (_Float16)t[c][r];
    lds_wave_fence();
    const v16h pa = frag_a_16x32(&Pt[wave][0][0], 32, lane);

    // O[16q x 64dk] += P[16x32] . V[32 x 64]; B from transposed V tile
#pragma unroll
    for (int j = 0; j < 4; ++j) {
      const v16h bvf = frag_b_from_T(&VtT[j * 16][0], 32, lane);
      o[j] = __builtin_amdgcn_wmma_f32_16x16x32_f16(false, pa, false, bvf,
                                                    (short)0, o[j], false, false);
    }
    __syncthreads();
  }

  // normalize + store f16 [b*N + q][h*64 + dk]
#pragma unroll
  for (int j = 0; j < 4; ++j) {
#pragma unroll
    for (int r = 0; r < 8; ++r) {
      const int q = qrowbase + r;
      const float v = o[j][r] / lrow[r];
      Oh[((size_t)(b * NN + q)) * DDIM + h * DKK + j * 16 + nloc] = (_Float16)v;
    }
  }
}

// ---------------------------------------------------------------------------
// host-side launcher
// ---------------------------------------------------------------------------
extern "C" void kernel_launch(void* const* d_in, const int* in_sizes, int n_in,
                              void* d_out, int out_size, void* d_ws, size_t ws_size,
                              hipStream_t stream) {
  (void)in_sizes; (void)n_in; (void)out_size; (void)ws_size;
  const float* Xq  = (const float*)d_in[0];
  const float* Xk  = (const float*)d_in[1];
  const float* Xv  = (const float*)d_in[2];
  const float* box = (const float*)d_in[3];
  const int*   msk = (const int*)d_in[4];
  const float* Wq = (const float*)d_in[5];
  const float* bq = (const float*)d_in[6];
  const float* Wk = (const float*)d_in[7];
  const float* bk = (const float*)d_in[8];
  const float* Wv = (const float*)d_in[9];
  const float* bv = (const float*)d_in[10];
  const float* Wo = (const float*)d_in[11];
  const float* bo = (const float*)d_in[12];
  const float* Wg = (const float*)d_in[13];
  const float* bg = (const float*)d_in[14];
  float* out = (float*)d_out;

  const size_t MN  = (size_t)BB * NN;   // 2048
  const size_t XSZ = MN * DDIM;         // 1,048,576 f16 elems
  const size_t WSZ = (size_t)DDIM * DDIM;

  _Float16* p = (_Float16*)d_ws;
  _Float16* Xq_h = p;            p += XSZ;
  _Float16* Xk_h = p;            p += XSZ;
  _Float16* Xv_h = p;            p += XSZ;
  _Float16* Wq_h = p;            p += WSZ;
  _Float16* Wk_h = p;            p += WSZ;
  _Float16* Wv_h = p;            p += WSZ;
  _Float16* Wo_h = p;            p += WSZ;
  _Float16* Qh   = p;            p += XSZ;
  _Float16* Kh   = p;            p += XSZ;
  _Float16* Vh   = p;            p += XSZ;
  _Float16* AOh  = p;            p += XSZ;
  float*    logg = (float*)p;    // 4*8*512*512 f32 = 32 MB

  auto cvt = [&](const float* s, _Float16* d, size_t n) {
    cvt_f16_kernel<<<dim3((unsigned)((n + 255) / 256)), dim3(256), 0, stream>>>(
        s, d, (int)n);
  };
  cvt(Xq, Xq_h, XSZ);
  cvt(Xk, Xk_h, XSZ);
  cvt(Xv, Xv_h, XSZ);
  cvt(Wq, Wq_h, WSZ);
  cvt(Wk, Wk_h, WSZ);
  cvt(Wv, Wv_h, WSZ);
  cvt(Wo, Wo_h, WSZ);

  const dim3 ggrid(DDIM / 64, (unsigned)(MN / 64));
  gemm_wmma_kernel<<<ggrid, dim3(128), 0, stream>>>(
      Xq_h, Wq_h, bq, Qh, nullptr, (int)MN, DDIM, DDIM);
  gemm_wmma_kernel<<<ggrid, dim3(128), 0, stream>>>(
      Xk_h, Wk_h, bk, Kh, nullptr, (int)MN, DDIM, DDIM);
  gemm_wmma_kernel<<<ggrid, dim3(128), 0, stream>>>(
      Xv_h, Wv_h, bv, Vh, nullptr, (int)MN, DDIM, DDIM);

  geo_kernel<<<dim3((unsigned)MN), dim3(256), 0, stream>>>(box, Wg, bg, msk, logg);

  attn_kernel<<<dim3(BB * HH * (NN / 64)), dim3(128), 0, stream>>>(
      Qh, Kh, Vh, logg, AOh);

  gemm_wmma_kernel<<<ggrid, dim3(128), 0, stream>>>(
      AOh, Wo_h, bo, nullptr, out, (int)MN, DDIM, DDIM);
}
